// SingleConv3DBlock_dsc_46608985096480
// MI455X (gfx1250) — matrix operand order repaired and checked by reading, not compiled, not run on hardware
//
#include <hip/hip_runtime.h>
#include <cstdint>

typedef __attribute__((ext_vector_type(16))) _Float16 v16h;
typedef __attribute__((ext_vector_type(8)))  float    v8f;

constexpr int Bsz = 2, Cin = 32, Cout = 64, Dd = 48, Hh = 48, Ww = 48, Kk = 3;
constexpr int HW  = Hh * Ww;            // 2304
constexpr int DHW = Dd * HW;            // 110592
constexpr int GROUPS = 16;
constexpr float EPS = 1e-5f;

// ---- workspace layout (float offsets) ----
constexpr size_t WS_RAW4  = 0;                                  // B*4*DHW (offset-conv, live chans {3,5,6,8})
constexpr size_t WS_STATS = WS_RAW4 + (size_t)Bsz * 4 * DHW;    // 8: sum[4], sumsq[4]
constexpr size_t WS_BN    = WS_STATS + 8;                       // 8: scale[4], shift[4]
constexpr size_t WS_GNRAW = WS_BN + 8;                          // B*16*2 = 64
constexpr size_t WS_GNFIN = WS_GNRAW + 64;                      // 64: mean, invstd

__device__ __forceinline__ int iclamp(int v, int lo, int hi) {
  return v < lo ? lo : (v > hi ? hi : v);
}

// ------------------------------------------------------------------
// 0) zero the stat accumulators (re-run every launch: deterministic)
// ------------------------------------------------------------------
__global__ void k_init(float* ws) {
  int i = threadIdx.x;
  if (i < 144) ws[WS_STATS + i] = 0.f;
}

// ------------------------------------------------------------------
// 1) offset conv: 3x3x3 SAME conv, only the 4 channels {3,5,6,8}
//    that feed the deformation; accumulate BN sum/sumsq.
// ------------------------------------------------------------------
__global__ __launch_bounds__(256) void k_offset_conv(
    const float* __restrict__ x, const float* __restrict__ ow,
    const float* __restrict__ ob, float* __restrict__ ws)
{
  __shared__ float wl[4 * 864];   // 4 chans * 32 cin * 27 taps
  __shared__ float bsum[8];
  const int chmap[4] = {3, 5, 6, 8};
  const int tid = threadIdx.x;
  for (int i = tid; i < 4 * 864; i += 256) {
    int c4 = i / 864, r = i - c4 * 864;
    wl[i] = ow[chmap[c4] * 864 + r];
  }
  if (tid < 8) bsum[tid] = 0.f;
  __syncthreads();

  size_t gid = (size_t)blockIdx.x * 256 + tid;     // over B*DHW (exact)
  int b  = (int)(gid / DHW);
  int sp = (int)(gid - (size_t)b * DHW);
  int d  = sp / HW;
  int hw = sp - d * HW;
  int h  = hw / Ww;
  int w  = hw - h * Ww;

  float a0 = ob[3], a1 = ob[5], a2 = ob[6], a3 = ob[8];
  const float* xb = x + (size_t)b * Cin * DHW;
  for (int c = 0; c < Cin; ++c) {
    const float* xc = xb + (size_t)c * DHW;
    const float* w0 = &wl[0 * 864 + c * 27];
    const float* w1 = &wl[1 * 864 + c * 27];
    const float* w2 = &wl[2 * 864 + c * 27];
    const float* w3 = &wl[3 * 864 + c * 27];
    int tap = 0;
    #pragma unroll
    for (int dz = -1; dz <= 1; ++dz) {
      int zz = d + dz;
      bool zok = (unsigned)zz < (unsigned)Dd;
      #pragma unroll
      for (int dy = -1; dy <= 1; ++dy) {
        int yy = h + dy;
        bool yok = zok && ((unsigned)yy < (unsigned)Hh);
        #pragma unroll
        for (int dx = -1; dx <= 1; ++dx, ++tap) {
          int xx = w + dx;
          float v = 0.f;
          if (yok && (unsigned)xx < (unsigned)Ww)
            v = xc[zz * HW + yy * Ww + xx];
          a0 = fmaf(v, w0[tap], a0);
          a1 = fmaf(v, w1[tap], a1);
          a2 = fmaf(v, w2[tap], a2);
          a3 = fmaf(v, w3[tap], a3);
        }
      }
    }
  }
  float* raw = ws + WS_RAW4;
  size_t pb = (size_t)b * 4 * DHW + sp;
  raw[pb + 0 * (size_t)DHW] = a0;
  raw[pb + 1 * (size_t)DHW] = a1;
  raw[pb + 2 * (size_t)DHW] = a2;
  raw[pb + 3 * (size_t)DHW] = a3;

  atomicAdd(&bsum[0], a0); atomicAdd(&bsum[4], a0 * a0);
  atomicAdd(&bsum[1], a1); atomicAdd(&bsum[5], a1 * a1);
  atomicAdd(&bsum[2], a2); atomicAdd(&bsum[6], a2 * a2);
  atomicAdd(&bsum[3], a3); atomicAdd(&bsum[7], a3 * a3);
  __syncthreads();
  if (tid < 8) atomicAdd(&ws[WS_STATS + tid], bsum[tid]);
}

// ------------------------------------------------------------------
// 2) fold batchnorm into per-channel scale/shift
// ------------------------------------------------------------------
__global__ void k_bn_fin(const float* __restrict__ bng,
                         const float* __restrict__ bnb, float* ws) {
  int c = threadIdx.x;
  if (c >= 4) return;
  const int chmap[4] = {3, 5, 6, 8};
  float N = (float)((size_t)Bsz * DHW);
  float mean = ws[WS_STATS + c] / N;
  float var  = ws[WS_STATS + 4 + c] / N - mean * mean;
  float sc   = bng[chmap[c]] * rsqrtf(var + EPS);
  ws[WS_BN + c]     = sc;
  ws[WS_BN + 4 + c] = bnb[chmap[c]] - mean * sc;
}

// ------------------------------------------------------------------
// 3) deformable sampling + (64 x 96) GEMM via v_wmma_f32_16x16x32_f16.
//    block = 96 threads = 3 waves; one block per (b,d,h) row, each wave
//    owns a 16-wide W tile. conv_w is staged to LDS with
//    GLOBAL_LOAD_ASYNC_TO_LDS_B128 (ASYNCcnt path); all 12 A-fragments
//    are then hoisted into registers so the inner loop is pure
//    gather -> cvt -> wmma. Reduction r = k*32 + cin: one WMMA K-step
//    fixes tap k, so one bilinear corner set serves 16 cin gathers.
// ------------------------------------------------------------------
__global__ __launch_bounds__(96) void k_deform_conv(
    const float* __restrict__ x, const float* __restrict__ cw,
    const float* __restrict__ cb, float* __restrict__ ws,
    float* __restrict__ out)
{
  __shared__ float wl[Cout * Cin * Kk];   // 6144 floats = 24.6 KB
  __shared__ float bl[Cout];
  const int tid = threadIdx.x;

  // ---- async stage conv_w -> LDS: 6144 floats = 1536 x b128,
  //      96 threads x 16 transfers each (GVS form: saddr + 32-bit voffs) ----
  {
    uint32_t lds_base = (uint32_t)(uintptr_t)(&wl[0]);  // low 32 bits = LDS addr
    #pragma unroll
    for (int i = 0; i < 16; ++i) {
      uint32_t boff = (uint32_t)(tid + i * 96) * 16u;   // byte offset
      uint32_t lds_addr = lds_base + boff;
      asm volatile("global_load_async_to_lds_b128 %0, %1, %2"
                   :: "v"(lds_addr), "v"(boff), "s"(cw)
                   : "memory");
    }
  }
  if (tid < Cout) bl[tid] = cb[tid];
  asm volatile("s_wait_asynccnt 0x0" ::: "memory");
  __syncthreads();

  const int lane = tid & 31;
  const int wave = tid >> 5;        // 0..2 -> W tile
  const int hi   = lane >> 4;       // half-wave
  const int pl   = lane & 15;       // position in tile (= N column)
  const int bdh  = blockIdx.x;
  const int h = bdh % Hh;
  const int d = (bdh / Hh) % Dd;
  const int b = bdh / (Hh * Dd);
  const int wpos = wave * 16 + pl;

  // ---- hoist all 12 A fragments (M x K = 16 x 32 f16) into registers.
  //      A layout: contiguous K = hi*16 + j (matches B fragment). ----
  v16h afrag[Kk][4];
  #pragma unroll
  for (int s = 0; s < Kk; ++s) {
    #pragma unroll
    for (int t = 0; t < 4; ++t) {
      int m = t * 16 + pl;
      #pragma unroll
      for (int j = 0; j < 16; ++j) {
        int kloc = hi * 16 + j;
        afrag[s][t][j] = (_Float16)wl[(m * Cin + kloc) * Kk + s]; // cin=kloc, tap=s
      }
    }
  }

  // tanh(batchnorm(.)) for the 4 live offset channels at this position
  const float* raw = ws + WS_RAW4;
  const float* bn  = ws + WS_BN;
  size_t rb = (size_t)b * 4 * DHW + (size_t)d * HW + h * Ww + wpos;
  float t0 = tanhf(fmaf(raw[rb              ], bn[0], bn[4]));  // cy[0]
  float t1 = tanhf(fmaf(raw[rb + 1ull * DHW], bn[1], bn[5]));   // cy[2]
  float t2 = tanhf(fmaf(raw[rb + 2ull * DHW], bn[2], bn[6]));   // cx[0]
  float t3 = tanhf(fmaf(raw[rb + 3ull * DHW], bn[3], bn[7]));   // cx[2]

  v8f Cacc[4] = {};   // 4 cout tiles of 16

  const float* xb = x + (size_t)b * Cin * DHW;
  const int cin_base = hi * 16;

  #pragma unroll
  for (int s = 0; s < Kk; ++s) {
    // deformed coords for tap s (z is integral -> bilinear in y,x)
    float cy = (s == 0) ? t0 : (s == 2) ? t1 : 0.f;
    float cx = (s == 0) ? t2 : (s == 2) ? t3 : 0.f;
    float yc = fminf(fmaxf((float)h + cy, 0.f), 47.f);
    float xc = fminf(fmaxf((float)wpos + cx, 0.f), 47.f);
    int y0 = (int)yc, x0 = (int)xc;           // trunc == floor (>=0)
    float fy = yc - (float)y0, fx = xc - (float)x0;
    int y1 = y0 + 1 > 47 ? 47 : y0 + 1;
    int x1 = x0 + 1 > 47 ? 47 : x0 + 1;
    float w00 = (1.f - fy) * (1.f - fx), w01 = (1.f - fy) * fx;
    float w10 = fy * (1.f - fx),         w11 = fy * fx;
    int z = iclamp(d + s - 1, 0, Dd - 1);
    const float* base = xb + (size_t)cin_base * DHW + (size_t)z * HW;
    int o00 = y0 * Ww + x0, o01 = y0 * Ww + x1;
    int o10 = y1 * Ww + x0, o11 = y1 * Ww + x1;

    // warm the two rows of the sampling footprint (global_prefetch_b8)
    __builtin_prefetch(base + o00, 0, 1);
    __builtin_prefetch(base + o10, 0, 1);

    // B fragment (K x N = 32 x 16 f16): lanes 0-15 hold K=0..15,
    // lanes 16-31 hold K=16..31; here K == cin.
    v16h bf;
    #pragma unroll
    for (int j = 0; j < 16; ++j) {
      const float* p = base + (size_t)j * DHW;
      float v = p[o00] * w00 + p[o01] * w01 + p[o10] * w10 + p[o11] * w11;
      bf[j] = (_Float16)v;
    }

    #pragma unroll
    for (int t = 0; t < 4; ++t) {
      Cacc[t] = __builtin_amdgcn_wmma_f32_16x16x32_f16(
          false, afrag[s][t], false, bf, (short)0, Cacc[t], false, false);
    }
  }

  // write pre-GN output + group stats.
  // C layout: VGPR v -> M = t*16 + hi*8 + v, column N = pl.
  size_t ob = (size_t)b * Cout * DHW + (size_t)d * HW + h * Ww + wpos;
  #pragma unroll
  for (int t = 0; t < 4; ++t) {
    float sA = 0.f, qA = 0.f, sB = 0.f, qB = 0.f;
    #pragma unroll
    for (int v = 0; v < 8; ++v) {
      int m = t * 16 + hi * 8 + v;
      float val = Cacc[t][v] + bl[m];
      out[ob + (size_t)m * DHW] = val;
      if (v < 4) { sA += val; qA += val * val; }
      else       { sB += val; qB += val * val; }
    }
    // reduce across the 16 positions (lane bits 0-3 stay in our half)
    #pragma unroll
    for (int mask = 1; mask < 16; mask <<= 1) {
      sA += __shfl_xor(sA, mask);
      qA += __shfl_xor(qA, mask);
      sB += __shfl_xor(sB, mask);
      qB += __shfl_xor(qB, mask);
    }
    if (pl == 0) {
      int g0 = t * 4 + hi * 2;                 // group = m/4
      atomicAdd(&ws[WS_GNRAW + ((size_t)b * GROUPS + g0) * 2 + 0], sA);
      atomicAdd(&ws[WS_GNRAW + ((size_t)b * GROUPS + g0) * 2 + 1], qA);
      atomicAdd(&ws[WS_GNRAW + ((size_t)b * GROUPS + g0 + 1) * 2 + 0], sB);
      atomicAdd(&ws[WS_GNRAW + ((size_t)b * GROUPS + g0 + 1) * 2 + 1], qB);
    }
  }
}

// ------------------------------------------------------------------
// 4) groupnorm stats -> mean, invstd
// ------------------------------------------------------------------
__global__ void k_gn_fin(float* ws) {
  int i = threadIdx.x;
  if (i >= Bsz * GROUPS) return;
  float N = (float)((size_t)(Cout / GROUPS) * DHW);   // 4*110592
  float s = ws[WS_GNRAW + (size_t)i * 2];
  float q = ws[WS_GNRAW + (size_t)i * 2 + 1];
  float mean = s / N;
  float var  = q / N - mean * mean;
  ws[WS_GNFIN + (size_t)i * 2]     = mean;
  ws[WS_GNFIN + (size_t)i * 2 + 1] = rsqrtf(var + EPS);
}

// ------------------------------------------------------------------
// 5) apply groupnorm + relu in place on d_out (float4 vectorized)
// ------------------------------------------------------------------
__global__ __launch_bounds__(256) void k_gn_apply(
    float* __restrict__ out, const float* __restrict__ ws,
    const float* __restrict__ gg, const float* __restrict__ gb)
{
  size_t idx = ((size_t)blockIdx.x * 256 + threadIdx.x) * 4;
  int c = (int)((idx / DHW) % Cout);
  int b = (int)(idx / ((size_t)Cout * DHW));
  int g = c >> 2;
  float mean = ws[WS_GNFIN + ((size_t)b * GROUPS + g) * 2];
  float inv  = ws[WS_GNFIN + ((size_t)b * GROUPS + g) * 2 + 1];
  float sc = inv * gg[c];
  float sh = gb[c] - mean * sc;
  float4 v = *(const float4*)(out + idx);
  v.x = fmaxf(fmaf(v.x, sc, sh), 0.f);
  v.y = fmaxf(fmaf(v.y, sc, sh), 0.f);
  v.z = fmaxf(fmaf(v.z, sc, sh), 0.f);
  v.w = fmaxf(fmaf(v.w, sc, sh), 0.f);
  *(float4*)(out + idx) = v;
}

// ------------------------------------------------------------------
extern "C" void kernel_launch(void* const* d_in, const int* in_sizes, int n_in,
                              void* d_out, int out_size, void* d_ws, size_t ws_size,
                              hipStream_t stream)
{
  const float* x   = (const float*)d_in[0];
  const float* ow  = (const float*)d_in[1];
  const float* ob  = (const float*)d_in[2];
  const float* bng = (const float*)d_in[3];
  const float* bnb = (const float*)d_in[4];
  const float* cw  = (const float*)d_in[5];
  const float* cb  = (const float*)d_in[6];
  const float* gg  = (const float*)d_in[7];
  const float* gb  = (const float*)d_in[8];
  float* out = (float*)d_out;
  float* ws  = (float*)d_ws;

  hipLaunchKernelGGL(k_init, dim3(1), dim3(256), 0, stream, ws);
  hipLaunchKernelGGL(k_offset_conv, dim3((Bsz * DHW) / 256), dim3(256), 0, stream,
                     x, ow, ob, ws);
  hipLaunchKernelGGL(k_bn_fin, dim3(1), dim3(32), 0, stream, bng, bnb, ws);
  hipLaunchKernelGGL(k_deform_conv, dim3(Bsz * Dd * Hh), dim3(96), 0, stream,
                     x, cw, cb, ws, out);
  hipLaunchKernelGGL(k_gn_fin, dim3(1), dim3(32), 0, stream, ws);
  hipLaunchKernelGGL(k_gn_apply, dim3((Bsz * Cout * DHW / 4) / 256), dim3(256), 0,
                     stream, out, ws, gg, gb);
}
